// SpeakerModel_1924145348860
// MI455X (gfx1250) — compile-verified, matrix operand order checked by reading, not run on hardware
//
#include <hip/hip_runtime.h>
#include <math.h>
#include <stdint.h>

// ---------------- model dimensions ----------------
#define BN   128
#define LPN  60
#define LUN  40
#define EN   512
#define HN   512
#define AN   512
#define IMGN 2048
#define VN   8000
#define G4H  2048   // 4*H

// ---------------- WMMA types ----------------
typedef __attribute__((ext_vector_type(16))) __bf16 bf16x16;
typedef __attribute__((ext_vector_type(8)))  float  f32x8;

union FragU {
  struct { uint4 lo, hi; } u;
  bf16x16 v;
};

__device__ __forceinline__ unsigned short f2bf(float f) {
  unsigned int u = __float_as_uint(f);
  unsigned int r = u + 0x7FFFu + ((u >> 16) & 1u);   // round-to-nearest-even
  return (unsigned short)(r >> 16);
}
__device__ __forceinline__ float sigf(float x) { return 1.f / (1.f + expf(-x)); }

// =====================================================================
// Generic bf16 GEMM with f32 accumulate:
//   C[m,n] (+)= sum_k A[m,k] * W[n,k]   (i.e. X @ W.T), optional bias+ReLU
// A: bf16 row-major, row stride lda (elements); W: bf16 row-major (N x K);
// C: f32, row stride ldc.
// Block = 4 wave32 waves (128 thr); block tile 16(M) x 128(N);
// wave tile 16 x 32 (two WMMA accumulators sharing one A fragment).
// A tile (16x32) double-buffered in LDS, staged with CDNA5
// GLOBAL_LOAD_ASYNC_TO_LDS_B128 (ASYNCcnt) so staging of the next K-slab
// overlaps the current WMMAs; one s_wait_asynccnt + one barrier per K-step.
// B fragments loaded per-lane from global with the 16-bit A/B VGPR layout
// (lane = N, K interleave 0-7/16-23 | 8-15/24-31 by lane half).
// =====================================================================
__global__ void __launch_bounds__(128) gemm_bf16_kernel(
    const unsigned short* __restrict__ A, int lda,
    const unsigned short* __restrict__ Bw,
    const float* __restrict__ bias,
    float* __restrict__ C, int ldc,
    int M, int N, int K, int accflag, int act)
{
  __shared__ unsigned short sA[2][16 * 40];   // 16 rows x 32 cols, +8 pad/row

  const int tid  = threadIdx.x;
  const int lane = tid & 31;
  const int wv   = tid >> 5;        // 0..3
  const int row  = lane & 15;       // M index (A frag) / N index (B frag, C col)
  const int hl   = lane >> 4;       // K-interleave half / +8 row half of C
  const int m0   = blockIdx.x * 16;
  const int col0 = blockIdx.y * 128 + wv * 32;
  const int n0   = col0 + row;
  const int n1   = n0 + 16;
  const bool v0  = (n0 < N), v1 = (n1 < N);
  const int nc0  = v0 ? n0 : (N - 1);
  const int nc1  = v1 ? n1 : (N - 1);

  f32x8 acc0 = {0.f,0.f,0.f,0.f,0.f,0.f,0.f,0.f};
  f32x8 acc1 = {0.f,0.f,0.f,0.f,0.f,0.f,0.f,0.f};
  if (accflag) {
    if (v0) {
#pragma unroll
      for (int r = 0; r < 8; ++r) acc0[r] = C[(size_t)(m0 + r + 8 * hl) * ldc + n0];
    }
    if (v1) {
#pragma unroll
      for (int r = 0; r < 8; ++r) acc1[r] = C[(size_t)(m0 + r + 8 * hl) * ldc + n1];
    }
  }

  const unsigned short* brow0 = Bw + (size_t)nc0 * K;
  const unsigned short* brow1 = Bw + (size_t)nc1 * K;

  // Staging roles: threads 0..63 (waves 0,1) each async-load one 16B chunk
  // of the 16x32 A tile: row = tid>>2, quarter = tid&3.
  const int sr   = tid >> 2;        // 0..15 (tid<64)
  const int part = tid & 3;         // 0..3
  const unsigned short* arow = A + (size_t)(m0 + sr) * lda + part * 8;

  // prologue: stage K-slab 0 into buffer 0
  if (tid < 64) {
    unsigned lds_off = (unsigned)(uintptr_t)&sA[0][sr * 40 + part * 8];
    unsigned long long ga = (unsigned long long)(uintptr_t)(arow);
    asm volatile("global_load_async_to_lds_b128 %0, %1, off"
                 :: "v"(lds_off), "v"(ga) : "memory");
  }
  asm volatile("s_wait_asynccnt 0x0" ::: "memory");
  __syncthreads();

  for (int kk = 0; kk < K; kk += 32) {
    const int cur = (kk >> 5) & 1;

    // stage next K-slab into the other buffer while we compute
    if (kk + 32 < K && tid < 64) {
      unsigned lds_off = (unsigned)(uintptr_t)&sA[cur ^ 1][sr * 40 + part * 8];
      unsigned long long ga = (unsigned long long)(uintptr_t)(arow + kk + 32);
      asm volatile("global_load_async_to_lds_b128 %0, %1, off"
                   :: "v"(lds_off), "v"(ga) : "memory");
    }

    FragU af, b0, b1;
    const unsigned short* ap = &sA[cur][row * 40 + hl * 8];
    af.u.lo = *(const uint4*)(ap);        // K = kk + hl*8 .. +7
    af.u.hi = *(const uint4*)(ap + 16);   // K = kk + 16 + hl*8 .. +7

    const unsigned short* bp0 = brow0 + kk + hl * 8;
    const unsigned short* bp1 = brow1 + kk + hl * 8;
    b0.u.lo = *(const uint4*)(bp0); b0.u.hi = *(const uint4*)(bp0 + 16);
    b1.u.lo = *(const uint4*)(bp1); b1.u.hi = *(const uint4*)(bp1 + 16);
    if (kk + 32 < K) {
      __builtin_prefetch(bp0 + 32, 0, 1);   // next K-slab of weights
      __builtin_prefetch(bp1 + 32, 0, 1);
    }

    acc0 = __builtin_amdgcn_wmma_f32_16x16x32_bf16(
        false, af.v, false, b0.v, (short)0, acc0, false, false);
    acc1 = __builtin_amdgcn_wmma_f32_16x16x32_bf16(
        false, af.v, false, b1.v, (short)0, acc1, false, false);

    asm volatile("s_wait_asynccnt 0x0" ::: "memory");  // next tile staged
    __syncthreads();
  }

  if (v0) {
    float bv = bias ? bias[n0] : 0.f;
#pragma unroll
    for (int r = 0; r < 8; ++r) {
      float v = acc0[r] + bv;
      if (act == 1) v = fmaxf(v, 0.f);
      C[(size_t)(m0 + r + 8 * hl) * ldc + n0] = v;
    }
  }
  if (v1) {
    float bv = bias ? bias[n1] : 0.f;
#pragma unroll
    for (int r = 0; r < 8; ++r) {
      float v = acc1[r] + bv;
      if (act == 1) v = fmaxf(v, 0.f);
      C[(size_t)(m0 + r + 8 * hl) * ldc + n1] = v;
    }
  }
}

// ---------------- elementwise helpers ----------------
__global__ void f2b_kernel(const float* __restrict__ src,
                           unsigned short* __restrict__ dst, long n) {
  long i = (long)blockIdx.x * blockDim.x + threadIdx.x;
  if (i < n) dst[i] = f2bf(src[i]);
}

__global__ void f2b_strided_kernel(const float* __restrict__ src,
                                   unsigned short* __restrict__ dst,
                                   int M, int Csrc, int dstStride, int colOff) {
  long i = (long)blockIdx.x * blockDim.x + threadIdx.x;
  if (i < (long)M * Csrc) {
    int r = (int)(i / Csrc), c = (int)(i % Csrc);
    dst[(size_t)r * dstStride + colOff + c] = f2bf(src[i]);
  }
}

__global__ void zero_u32_kernel(unsigned int* __restrict__ p, long n) {
  long i = (long)blockIdx.x * blockDim.x + threadIdx.x;
  if (i < n) p[i] = 0u;
}

__global__ void copy_state_kernel(const float* __restrict__ src,
                                  float* __restrict__ h, float* __restrict__ c,
                                  unsigned short* __restrict__ hbf, int n) {
  int i = blockIdx.x * blockDim.x + threadIdx.x;
  if (i < n) { float v = src[i]; h[i] = v; c[i] = v; hbf[i] = f2bf(v); }
}

// Embedding gathers (fp32 table -> bf16 activations), forward + time-reversed.
__global__ void gather_x_kernel(const float* __restrict__ emb,
                                const int* __restrict__ toks,
                                const int* __restrict__ lens,
                                unsigned short* __restrict__ x,
                                unsigned short* __restrict__ xrev) {
  long idx = (long)blockIdx.x * blockDim.x + threadIdx.x;
  if (idx >= (long)BN * LPN * EN) return;
  int j = (int)(idx % EN);
  long bt = idx / EN;
  int t = (int)(bt % LPN), b = (int)(bt / LPN);
  int tok = toks[b * LPN + t];
  x[idx] = f2bf(emb[(size_t)tok * EN + j]);
  int rt = lens[b] - 1 - t; if (rt < 0) rt = 0;
  int tok2 = toks[b * LPN + rt];
  xrev[idx] = f2bf(emb[(size_t)tok2 * EN + j]);
}

__global__ void gather_u_kernel(const float* __restrict__ emb,
                                const int* __restrict__ toks,
                                unsigned short* __restrict__ xu) {
  long idx = (long)blockIdx.x * blockDim.x + threadIdx.x;
  if (idx >= (long)BN * LUN * EN) return;
  int j = (int)(idx % EN);
  long bt = idx / EN;
  int t = (int)(bt % LUN), b = (int)(bt / LUN);
  int tok = toks[b * LUN + t];
  xu[idx] = f2bf(emb[(size_t)tok * EN + j]);
}

// LSTM gate nonlinearity + masked state update.
// dir 0: forward encoder  (writes outputs[b,t,0:512] = m ? h' : 0)
// dir 1: backward encoder (writes outputs[b,len-1-t,512:1024] = h' when valid)
// dir 2: decoder          (no mask, no output buffer)
__global__ void lstm_pointwise_kernel(const float* __restrict__ g,
                                      float* __restrict__ h, float* __restrict__ c,
                                      unsigned short* __restrict__ hbf,
                                      unsigned short* __restrict__ outbuf,
                                      const int* __restrict__ lens, int t, int dir) {
  int idx = blockIdx.x * blockDim.x + threadIdx.x;
  if (idx >= BN * HN) return;
  int b = idx / HN, j = idx % HN;
  const float* gb = g + (size_t)b * G4H;
  float gi = sigf(gb[j]);
  float gf = sigf(gb[HN + j]);
  float gg = tanhf(gb[2 * HN + j]);
  float go = sigf(gb[3 * HN + j]);
  float cn = gf * c[idx] + gi * gg;
  float hn = go * tanhf(cn);
  bool m = true;
  int len = 0;
  if (dir < 2) { len = lens[b]; m = (t < len); }
  float hq = m ? hn : h[idx];
  float cq = m ? cn : c[idx];
  h[idx] = hq; c[idx] = cq;
  hbf[idx] = f2bf(hq);
  if (dir == 0) {
    outbuf[((size_t)b * LPN + t) * (2 * HN) + j] = m ? f2bf(hn) : (unsigned short)0;
  } else if (dir == 1) {
    if (m) outbuf[((size_t)b * LPN + (len - 1 - t)) * (2 * HN) + HN + j] = f2bf(hn);
  }
}

// Attention: scores = tanh(hist + h_att) @ W_att + b_att, mask t>=len with -inf,
// softmax over t, ctx = sum_t w[t] * hist[b,t,:]. Emits [h, ctx] as bf16 concat.
__global__ void __launch_bounds__(256) attention_kernel(
    const float* __restrict__ hist,   // (B, LP, A)
    const float* __restrict__ hatt,   // (B, A)
    const float* __restrict__ h,      // (B, H) decoder state (fp32)
    const float* __restrict__ Watt,   // (A)
    const float* __restrict__ batt,   // (1)
    const int* __restrict__ lens,     // (B)
    unsigned short* __restrict__ cat) // (B, 2H) bf16 : [h | ctx]
{
  __shared__ float s_sc[64];
  __shared__ float s_inv;
  int b = blockIdx.x;
  int tid = threadIdx.x, lane = tid & 31, wv = tid >> 5;
  int len = lens[b];
  const float* hb = hist + (size_t)b * LPN * AN;
  const float* ha = hatt + (size_t)b * AN;

  for (int t = wv; t < LPN; t += 8) {
    float p = 0.f;
    const float* ht = hb + (size_t)t * AN;
    for (int j = lane; j < AN; j += 32)
      p += tanhf(ht[j] + ha[j]) * Watt[j];
#pragma unroll
    for (int off = 16; off; off >>= 1) p += __shfl_xor(p, off, 32);
    if (lane == 0) s_sc[t] = (t < len) ? (p + batt[0]) : -__builtin_inff();
  }
  __syncthreads();
  if (tid == 0) {
    float mx = -__builtin_inff();
    for (int t = 0; t < LPN; ++t) mx = fmaxf(mx, s_sc[t]);
    float sum = 0.f;
    for (int t = 0; t < LPN; ++t) { float e = expf(s_sc[t] - mx); s_sc[t] = e; sum += e; }
    s_inv = 1.f / sum;
  }
  __syncthreads();
  float inv = s_inv;
  for (int j = tid; j < AN; j += blockDim.x) {
    float accv = 0.f;
    for (int t = 0; t < len; ++t) accv += s_sc[t] * hb[(size_t)t * AN + j];
    accv *= inv;
    cat[(size_t)b * (2 * HN) + HN + j] = f2bf(accv);
    cat[(size_t)b * (2 * HN) + j]      = f2bf(h[(size_t)b * HN + j]);
  }
}

// =====================================================================
// Host-side orchestration
// =====================================================================
extern "C" void kernel_launch(void* const* d_in, const int* in_sizes, int n_in,
                              void* d_out, int out_size, void* d_ws, size_t ws_size,
                              hipStream_t stream)
{
  // ---- inputs (setup_inputs dict order) ----
  const int*   utterance = (const int*)d_in[0];
  const int*   prev_utt  = (const int*)d_in[1];
  const int*   lens      = (const int*)d_in[2];
  // d_in[3] = masks (recomputed from lens; dtype avoided)
  const float* visual    = (const float*)d_in[4];
  const float* tif       = (const float*)d_in[5];
  const float* emb       = (const float*)d_in[6];
  const float* enc_Wih_f = (const float*)d_in[7];
  const float* enc_Whh_f = (const float*)d_in[8];
  const float* enc_bih_f = (const float*)d_in[9];
  const float* enc_bhh_f = (const float*)d_in[10];
  const float* enc_Wih_b = (const float*)d_in[11];
  const float* enc_Whh_b = (const float*)d_in[12];
  const float* enc_bih_b = (const float*)d_in[13];
  const float* enc_bhh_b = (const float*)d_in[14];
  const float* dec_Wih   = (const float*)d_in[15];
  const float* dec_Whh   = (const float*)d_in[16];
  const float* dec_bih   = (const float*)d_in[17];
  const float* dec_bhh   = (const float*)d_in[18];
  const float* W_hid     = (const float*)d_in[19];
  const float* b_hid     = (const float*)d_in[20];
  const float* W_dec     = (const float*)d_in[21];
  const float* b_dec     = (const float*)d_in[22];
  const float* W_vc      = (const float*)d_in[23];
  const float* b_vc      = (const float*)d_in[24];
  const float* W_sep     = (const float*)d_in[25];
  const float* b_sep     = (const float*)d_in[26];
  const float* W2ah      = (const float*)d_in[27];
  const float* b2ah      = (const float*)d_in[28];
  const float* W2ahist   = (const float*)d_in[29];
  const float* b2ahist   = (const float*)d_in[30];
  const float* W_voc     = (const float*)d_in[31];
  const float* b_voc     = (const float*)d_in[32];
  const float* W_att     = (const float*)d_in[33];
  const float* b_att     = (const float*)d_in[34];
  float* out = (float*)d_out;

  // ---- workspace carve ----
  char* wp = (char*)d_ws;
  auto alloc = [&](size_t bytes) -> void* {
    void* r = (void*)wp;
    wp += (bytes + 255) & ~(size_t)255;
    return r;
  };
  auto ab = [&](size_t elems) { return (unsigned short*)alloc(elems * 2); }; // bf16
  auto af = [&](size_t elems) { return (float*)alloc(elems * 4); };          // fp32

  // bf16 weight copies (resident in L2 after first use)
  unsigned short* wWihF  = ab((size_t)G4H * EN);
  unsigned short* wWhhF  = ab((size_t)G4H * HN);
  unsigned short* wWihB  = ab((size_t)G4H * EN);
  unsigned short* wWhhB  = ab((size_t)G4H * HN);
  unsigned short* wDWih  = ab((size_t)G4H * EN);
  unsigned short* wDWhh  = ab((size_t)G4H * HN);
  unsigned short* wHid   = ab((size_t)HN * 2 * HN);
  unsigned short* wDec   = ab((size_t)HN * 2 * HN);
  unsigned short* wVc    = ab((size_t)HN * 6 * IMGN);
  unsigned short* wSep   = ab((size_t)HN * IMGN);
  unsigned short* w2ah   = ab((size_t)AN * HN);
  unsigned short* w2ahist= ab((size_t)AN * 2 * HN);
  unsigned short* wVoc   = ab((size_t)VN * (AN + HN));
  // bf16 activations
  unsigned short* viscB  = ab((size_t)BN * 6 * IMGN);
  unsigned short* tifB   = ab((size_t)BN * IMGN);
  unsigned short* xB     = ab((size_t)BN * LPN * EN);
  unsigned short* xrevB  = ab((size_t)BN * LPN * EN);
  unsigned short* embuB  = ab((size_t)BN * LUN * EN);
  unsigned short* outsB  = ab((size_t)BN * LPN * 2 * HN);   // encoder outputs [fwd|bwd]
  unsigned short* cat1   = ab((size_t)BN * 2 * HN);         // reused concat buffer
  unsigned short* hbf    = ab((size_t)BN * HN);
  // fp32 buffers
  float* vis_hid = af((size_t)BN * HN);
  float* tgt_hid = af((size_t)BN * HN);
  float* cvi     = af((size_t)BN * HN);
  float* dech    = af((size_t)BN * HN);
  float* gbuf    = af((size_t)BN * G4H);
  float* hst     = af((size_t)BN * HN);
  float* cst     = af((size_t)BN * HN);
  float* hist    = af((size_t)BN * LPN * AN);
  float* hatt    = af((size_t)BN * AN);

  auto f2b = [&](const float* s, unsigned short* d, long n) {
    f2b_kernel<<<(unsigned)((n + 255) / 256), 256, 0, stream>>>(s, d, n);
  };
  auto gemm = [&](const unsigned short* A, int lda, const unsigned short* Bw,
                  const float* bias, float* C, int ldc,
                  int M, int N, int K, int accf, int act) {
    dim3 g((unsigned)(M / 16), (unsigned)((N + 127) / 128));
    gemm_bf16_kernel<<<g, 128, 0, stream>>>(A, lda, Bw, bias, C, ldc, M, N, K, accf, act);
  };

  // ---- weight / input conversions ----
  f2b(enc_Wih_f, wWihF, (long)G4H * EN);
  f2b(enc_Whh_f, wWhhF, (long)G4H * HN);
  f2b(enc_Wih_b, wWihB, (long)G4H * EN);
  f2b(enc_Whh_b, wWhhB, (long)G4H * HN);
  f2b(dec_Wih,   wDWih, (long)G4H * EN);
  f2b(dec_Whh,   wDWhh, (long)G4H * HN);
  f2b(W_hid,     wHid,  (long)HN * 2 * HN);
  f2b(W_dec,     wDec,  (long)HN * 2 * HN);
  f2b(W_vc,      wVc,   (long)HN * 6 * IMGN);
  f2b(W_sep,     wSep,  (long)HN * IMGN);
  f2b(W2ah,      w2ah,  (long)AN * HN);
  f2b(W2ahist,   w2ahist, (long)AN * 2 * HN);
  f2b(W_voc,     wVoc,  (long)VN * (AN + HN));
  f2b(visual,    viscB, (long)BN * 6 * IMGN);
  f2b(tif,       tifB,  (long)BN * IMGN);

  // zero encoder output buffer (backward half only written at valid slots)
  {
    long n32 = (long)BN * LPN * 2 * HN / 2;   // u16 count / 2 = u32 count
    zero_u32_kernel<<<(unsigned)((n32 + 255) / 256), 256, 0, stream>>>(
        (unsigned int*)outsB, n32);
  }

  // ---- visual pathway ----
  gemm(viscB, 6 * IMGN, wVc,  b_vc,  vis_hid, HN, BN, HN, 6 * IMGN, 0, 1);
  gemm(tifB,  IMGN,     wSep, b_sep, tgt_hid, HN, BN, HN, IMGN,     0, 1);
  {
    long n = (long)BN * HN;
    unsigned bl = (unsigned)((n + 255) / 256);
    f2b_strided_kernel<<<bl, 256, 0, stream>>>(vis_hid, cat1, BN, HN, 2 * HN, 0);
    f2b_strided_kernel<<<bl, 256, 0, stream>>>(tgt_hid, cat1, BN, HN, 2 * HN, HN);
  }
  gemm(cat1, 2 * HN, wHid, b_hid, cvi, HN, BN, HN, 2 * HN, 0, 1);

  // ---- embedding gathers ----
  {
    long n = (long)BN * LPN * EN;
    gather_x_kernel<<<(unsigned)((n + 255) / 256), 256, 0, stream>>>(
        emb, prev_utt, lens, xB, xrevB);
    long nu = (long)BN * LUN * EN;
    gather_u_kernel<<<(unsigned)((nu + 255) / 256), 256, 0, stream>>>(
        emb, utterance, embuB);
  }

  const unsigned stBlocks = (BN * HN + 255) / 256;

  // ---- forward encoder LSTM ----
  copy_state_kernel<<<stBlocks, 256, 0, stream>>>(cvi, hst, cst, hbf, BN * HN);
  for (int t = 0; t < LPN; ++t) {
    gemm(xB + (size_t)t * EN, LPN * EN, wWihF, enc_bih_f, gbuf, G4H, BN, G4H, EN, 0, 0);
    gemm(hbf, HN,             wWhhF, enc_bhh_f, gbuf, G4H, BN, G4H, HN, 1, 0);
    lstm_pointwise_kernel<<<stBlocks, 256, 0, stream>>>(
        gbuf, hst, cst, hbf, outsB, lens, t, 0);
  }
  {
    long n = (long)BN * HN;
    f2b_strided_kernel<<<(unsigned)((n + 255) / 256), 256, 0, stream>>>(
        hst, cat1, BN, HN, 2 * HN, 0);     // h_f -> concat[:, 0:H]
  }

  // ---- backward encoder LSTM ----
  copy_state_kernel<<<stBlocks, 256, 0, stream>>>(cvi, hst, cst, hbf, BN * HN);
  for (int t = 0; t < LPN; ++t) {
    gemm(xrevB + (size_t)t * EN, LPN * EN, wWihB, enc_bih_b, gbuf, G4H, BN, G4H, EN, 0, 0);
    gemm(hbf, HN,                wWhhB, enc_bhh_b, gbuf, G4H, BN, G4H, HN, 1, 0);
    lstm_pointwise_kernel<<<stBlocks, 256, 0, stream>>>(
        gbuf, hst, cst, hbf, outsB, lens, t, 1);
  }
  {
    long n = (long)BN * HN;
    f2b_strided_kernel<<<(unsigned)((n + 255) / 256), 256, 0, stream>>>(
        hst, cat1, BN, HN, 2 * HN, HN);    // h_b -> concat[:, H:2H]
  }

  // ---- decoder init + history projection ----
  gemm(cat1, 2 * HN, wDec, b_dec, dech, HN, BN, HN, 2 * HN, 0, 0);
  copy_state_kernel<<<stBlocks, 256, 0, stream>>>(dech, hst, cst, hbf, BN * HN);
  gemm(outsB, 2 * HN, w2ahist, b2ahist, hist, AN, BN * LPN, AN, 2 * HN, 0, 0);

  // ---- decoder loop ----
  for (int t = 0; t < LUN - 1; ++t) {
    gemm(embuB + (size_t)t * EN, LUN * EN, wDWih, dec_bih, gbuf, G4H, BN, G4H, EN, 0, 0);
    gemm(hbf, HN,                wDWhh, dec_bhh, gbuf, G4H, BN, G4H, HN, 1, 0);
    lstm_pointwise_kernel<<<stBlocks, 256, 0, stream>>>(
        gbuf, hst, cst, hbf, (unsigned short*)nullptr, lens, t, 2);
    gemm(hbf, HN, w2ah, b2ah, hatt, AN, BN, AN, HN, 0, 0);
    attention_kernel<<<BN, 256, 0, stream>>>(hist, hatt, hst, W_att, b_att, lens, cat1);
    // preds[b, t, :]  (output layout (B, LU-1, V))
    gemm(cat1, 2 * HN, wVoc, b_voc, out + (size_t)t * VN, (LUN - 1) * VN,
         BN, VN, AN + HN, 0, 0);
  }
}
